// InforMax_78563541779006
// MI455X (gfx1250) — matrix operand order repaired and checked by reading, compiled, not run on hardware
//
#include <hip/hip_runtime.h>
#include <hip/hip_bf16.h>

typedef __attribute__((ext_vector_type(16))) _Float16 v16h;
typedef __attribute__((ext_vector_type(8)))  _Float16 v8h;
typedef __attribute__((ext_vector_type(8)))  float    v8f;

#define NUM_U 100000
#define NUM_I 50000
#define NNZE  1600000
#define BATCH 8192
#define NEG_SLOPE 0.2f

static __device__ __forceinline__ float leaky(float x) {
    return x > 0.0f ? x : NEG_SLOPE * x;
}

// A fragment: 16x32 f16 tile (M x K), row-major source, leading dim `ld`.
// Contiguous K order: lanes 0-15 -> M=lane, halves 0..15 = K 0..15;
// lanes 16-31 -> M=lane-16, K 16..31 (matches load_b_swz).
// Each half-run is 16 contiguous bytes -> 2x ds_load_b128.
static __device__ __forceinline__ v16h load_a_frag(const _Float16* src, int ld,
                                                   int kbase, int lane) {
    const int h = lane >> 4;
    const int m = lane & 15;
    const _Float16* row = src + m * ld + kbase + h * 16;
    v8h lo = *(const v8h*)(row);
    v8h hi = *(const v8h*)(row + 8);
    return __builtin_shufflevector(lo, hi, 0, 1, 2, 3, 4, 5, 6, 7,
                                   8, 9, 10, 11, 12, 13, 14, 15);
}

// B fragment from swizzled weights. Swizzled layout: [k>>4][n][k&15], i.e.
// the 16 K-values a lane needs (fixed column, one 16-row K chunk) are
// contiguous -> 2x ds_load_b128.
// Fragment: lanes 0-15 hold K=kbase..kbase+15 (N=lane), lanes 16-31 hold
// K=kbase+16..kbase+31.
static __device__ __forceinline__ v16h load_b_swz(const _Float16* wt, int N,
                                                  int kbase, int nbase, int lane) {
    const int h  = lane >> 4;
    const int nn = lane & 15;
    const _Float16* p = wt + ((((kbase >> 4) + h) * N) + (nbase + nn)) * 16;
    v8h lo = *(const v8h*)(p);
    v8h hi = *(const v8h*)(p + 8);
    return __builtin_shufflevector(lo, hi, 0, 1, 2, 3, 4, 5, 6, 7,
                                   8, 9, 10, 11, 12, 13, 14, 15);
}

static __device__ __forceinline__ v8f wmma_f16(v16h a, v16h b, v8f c) {
    return __builtin_amdgcn_wmma_f32_16x16x32_f16(false, a, false, b,
                                                  (short)0, c, false, false);
}

// ---------------------------------------------------------------------------
// Fused AttributeFilter forward: Y = leaky(leaky(X@W1 + b1)@W2 + b2)
// X:[n,64] f32, W1:[64,128], W2:[128,64]. 128 threads = 4 waves, 64 rows/block.
// ---------------------------------------------------------------------------
__global__ __launch_bounds__(128)
void filter_fwd_kernel(const float* __restrict__ X, int n,
                       const float* __restrict__ W1, const float* __restrict__ B1,
                       const float* __restrict__ W2, const float* __restrict__ B2,
                       float* __restrict__ Y) {
    __shared__ __align__(32) _Float16 sW1[64 * 128];   // swizzled [k>>4][n][k&15]
    __shared__ __align__(32) _Float16 sW2[128 * 64];   // swizzled
    __shared__ float    sB1[128];
    __shared__ float    sB2[64];
    __shared__ __align__(32) _Float16 sX[64 * 64];     // row-major tile
    __shared__ __align__(32) _Float16 sH[64 * 128];    // row-major per-wave H

    const int tid      = threadIdx.x;
    const int wave     = tid >> 5;
    const int lane     = tid & 31;
    const int blockRow = blockIdx.x * 64;

    for (int i = tid; i < 64 * 128; i += 128) {        // W1: K=64, N=128
        int k = i >> 7, nc = i & 127;
        sW1[(((k >> 4) * 128) + nc) * 16 + (k & 15)] = (_Float16)W1[i];
    }
    for (int i = tid; i < 128 * 64; i += 128) {        // W2: K=128, N=64
        int k = i >> 6, nc = i & 63;
        sW2[(((k >> 4) * 64) + nc) * 16 + (k & 15)] = (_Float16)W2[i];
    }
    if (tid < 128) sB1[tid] = B1[tid];
    if (tid < 64)  sB2[tid] = B2[tid];
    for (int i = tid; i < 64 * 64; i += 128) {
        int r = i >> 6, c = i & 63;
        int g = blockRow + r;
        sX[i] = (g < n) ? (_Float16)X[(long long)g * 64 + c] : (_Float16)0.0f;
    }
    __syncthreads();

    const _Float16* xt = sX + wave * 16 * 64;
    _Float16*       ht = sH + wave * 16 * 128;
    const int nn = lane & 15;
    const int mh = (lane >> 4) * 8;

    // Stage 1: H = leaky(X@W1 + b1), 16x128 per wave.  A frags hoisted.
    v16h a0 = load_a_frag(xt, 64, 0, lane);
    v16h a1 = load_a_frag(xt, 64, 32, lane);
#pragma unroll
    for (int nt = 0; nt < 8; ++nt) {
        v8f c = {};
        c = wmma_f16(a0, load_b_swz(sW1, 128, 0, nt * 16, lane), c);
        c = wmma_f16(a1, load_b_swz(sW1, 128, 32, nt * 16, lane), c);
        float bias = sB1[nt * 16 + nn];
#pragma unroll
        for (int r = 0; r < 8; ++r) {
            float v = leaky(c[r] + bias);
            ht[(mh + r) * 128 + nt * 16 + nn] = (_Float16)v;
        }
    }
    // same-wave LDS ops are in-order; compiler inserts dscnt waits

    // Stage 2: Y = leaky(H@W2 + b2), 16x64 per wave.  A frags hoisted.
    v16h h0 = load_a_frag(ht, 128, 0, lane);
    v16h h1 = load_a_frag(ht, 128, 32, lane);
    v16h h2 = load_a_frag(ht, 128, 64, lane);
    v16h h3 = load_a_frag(ht, 128, 96, lane);
#pragma unroll
    for (int nt = 0; nt < 4; ++nt) {
        v8f c = {};
        c = wmma_f16(h0, load_b_swz(sW2, 64, 0, nt * 16, lane), c);
        c = wmma_f16(h1, load_b_swz(sW2, 64, 32, nt * 16, lane), c);
        c = wmma_f16(h2, load_b_swz(sW2, 64, 64, nt * 16, lane), c);
        c = wmma_f16(h3, load_b_swz(sW2, 64, 96, nt * 16, lane), c);
        float bias = sB2[nt * 16 + nn];
#pragma unroll
        for (int r = 0; r < 8; ++r) {
            int g = blockRow + wave * 16 + mh + r;
            if (g < n)
                Y[(long long)g * 64 + nt * 16 + nn] = leaky(c[r] + bias);
        }
    }
}

// ---------------------------------------------------------------------------
// Discriminator 0 (nc=2): H = leaky(X@W1 + b1); logits = H@W2 + b2;
// accumulate sum of per-row cross-entropy into *acc.
// labels = users_features[:,0] (stride-3 int32).
// ---------------------------------------------------------------------------
__global__ __launch_bounds__(128)
void disc_loss_kernel(const float* __restrict__ X, int n,
                      const float* __restrict__ W1, const float* __restrict__ B1,
                      const float* __restrict__ W2, const float* __restrict__ B2,
                      const int* __restrict__ feats, float* __restrict__ acc) {
    __shared__ __align__(32) _Float16 sW1[64 * 64];    // swizzled
    __shared__ float    sB1[64];
    __shared__ float    sW2[128];   // [64][2]
    __shared__ float    sB2[2];
    __shared__ __align__(32) _Float16 sX[64 * 64];
    __shared__ float    sH[64 * 64];
    __shared__ float    red[128];

    const int tid      = threadIdx.x;
    const int wave     = tid >> 5;
    const int lane     = tid & 31;
    const int blockRow = blockIdx.x * 64;

    for (int i = tid; i < 64 * 64; i += 128) {         // W1: K=64, N=64
        int k = i >> 6, nc = i & 63;
        sW1[(((k >> 4) * 64) + nc) * 16 + (k & 15)] = (_Float16)W1[i];
    }
    if (tid < 64)  sB1[tid] = B1[tid];
    if (tid < 128) sW2[tid] = W2[tid];
    if (tid < 2)   sB2[tid] = B2[tid];
    for (int i = tid; i < 64 * 64; i += 128) {
        int r = i >> 6, c = i & 63;
        int g = blockRow + r;
        sX[i] = (g < n) ? (_Float16)X[(long long)g * 64 + c] : (_Float16)0.0f;
    }
    __syncthreads();

    const _Float16* xt = sX + wave * 16 * 64;
    float*          ht = sH + wave * 16 * 64;
    const int nn = lane & 15;
    const int mh = (lane >> 4) * 8;

    v16h a0 = load_a_frag(xt, 64, 0, lane);
    v16h a1 = load_a_frag(xt, 64, 32, lane);
#pragma unroll
    for (int nt = 0; nt < 4; ++nt) {
        v8f c = {};
        c = wmma_f16(a0, load_b_swz(sW1, 64, 0, nt * 16, lane), c);
        c = wmma_f16(a1, load_b_swz(sW1, 64, 32, nt * 16, lane), c);
        float bias = sB1[nt * 16 + nn];
#pragma unroll
        for (int r = 0; r < 8; ++r)
            ht[(mh + r) * 64 + nt * 16 + nn] = leaky(c[r] + bias);
    }
    __syncthreads();

    float loss = 0.0f;
    if (tid < 64) {
        int g = blockRow + tid;
        if (g < n) {
            const float* hrow = sH + tid * 64;
            float l0 = sB2[0], l1 = sB2[1];
#pragma unroll 8
            for (int k = 0; k < 64; ++k) {
                float h = hrow[k];
                l0 += h * sW2[k * 2 + 0];
                l1 += h * sW2[k * 2 + 1];
            }
            float mx  = fmaxf(l0, l1);
            float lse = mx + logf(expf(l0 - mx) + expf(l1 - mx));
            int lab   = feats[(long long)g * 3];
            loss = lse - (lab == 0 ? l0 : l1);
        }
    }
    red[tid] = loss;
    __syncthreads();
#pragma unroll
    for (int s = 64; s > 0; s >>= 1) {
        if (tid < s) red[tid] += red[tid + s];
        __syncthreads();
    }
    if (tid == 0) atomicAdd(acc, red[0]);
}

// ---------------------------------------------------------------------------
// COO SpMM: out[row,:] += val * feat[col,:].  One wave per edge, 2 floats/lane.
// ---------------------------------------------------------------------------
__global__ __launch_bounds__(256)
void spmm_kernel(const int* __restrict__ rows, const int* __restrict__ cols,
                 const float* __restrict__ vals, const float* __restrict__ feat,
                 float* __restrict__ out) {
    long long t = (long long)blockIdx.x * blockDim.x + threadIdx.x;
    int e    = (int)(t >> 5);
    int lane = threadIdx.x & 31;
    if (e >= NNZE) return;
    if (lane == 0 && e + 4096 < NNZE) {     // global_prefetch_b8 for edge stream
        __builtin_prefetch(rows + e + 4096, 0, 0);
        __builtin_prefetch(cols + e + 4096, 0, 0);
        __builtin_prefetch(vals + e + 4096, 0, 0);
    }
    int   r = rows[e];
    int   c = cols[e];
    float v = vals[e];
    const float2 f = *(const float2*)(feat + (long long)c * 64 + lane * 2);
    float* dst = out + (long long)r * 64 + lane * 2;
    atomicAdd(dst + 0, v * f.x);
    atomicAdd(dst + 1, v * f.y);
}

// ---------------------------------------------------------------------------
// Rating branch: accumulate sum((pred-r)^2) -> acc[2], sum(0.01*(|u|^2+|v|^2))
// -> acc[3].  One wave per batch element.
// ---------------------------------------------------------------------------
__global__ __launch_bounds__(256)
void pred_kernel(const float* __restrict__ uF, const float* __restrict__ iF,
                 const int* __restrict__ ub, const int* __restrict__ ib,
                 const float* __restrict__ rb, float* __restrict__ acc) {
    int wid  = (blockIdx.x * blockDim.x + threadIdx.x) >> 5;
    int lane = threadIdx.x & 31;
    if (wid >= BATCH) return;
    const float* u = uF + (long long)ub[wid] * 64;
    const float* v = iF + (long long)ib[wid] * 64;
    float u0 = u[lane], u1 = u[lane + 32];
    float v0 = v[lane], v1 = v[lane + 32];
    float dot = u0 * v0 + u1 * v1;
    float l2  = u0 * u0 + u1 * u1 + v0 * v0 + v1 * v1;
#pragma unroll
    for (int off = 16; off > 0; off >>= 1) {
        dot += __shfl_xor(dot, off, 32);
        l2  += __shfl_xor(l2, off, 32);
    }
    if (lane == 0) {
        float d = dot - rb[wid];
        atomicAdd(acc + 2, d * d);
        atomicAdd(acc + 3, 0.01f * l2);
    }
}

__global__ void zero_kernel(float* __restrict__ p, long long nfl) {
    long long i = (long long)blockIdx.x * blockDim.x + threadIdx.x;
    if (i < nfl) p[i] = 0.0f;
}

__global__ void finalize_kernel(const float* __restrict__ acc, float* __restrict__ out) {
    if (threadIdx.x == 0 && blockIdx.x == 0) {
        float d_loss       = acc[0] / (float)NUM_U;
        float d_loss_local = acc[1] / (float)NUM_U;
        float d_all        = d_loss + 0.5f * d_loss_local;   // W_F[0]=1, mask [1,0,0]
        float g_all        = 10.0f * (acc[2] / (float)BATCH + acc[3] / (float)BATCH);
        out[0] = d_all;
        out[1] = g_all;
        out[2] = -1000.0f * d_all;
    }
}

extern "C" void kernel_launch(void* const* d_in, const int* in_sizes, int n_in,
                              void* d_out, int out_size, void* d_ws, size_t ws_size,
                              hipStream_t stream) {
    // Input order (flattened setup_inputs dict):
    // 0 user_emb, 1 item_emb, 2..13 filters (3x W1,b1,W2,b2),
    // 14..25 discs (3x W1,b1,W2,b2), 26 rows, 27 cols, 28 vals,
    // 29 users_features, 30 user_batch, 31 rating_batch, 32 item_batch
    const float* user_emb = (const float*)d_in[0];
    const float* item_emb = (const float*)d_in[1];
    const float* fW1 = (const float*)d_in[2];
    const float* fB1 = (const float*)d_in[3];
    const float* fW2 = (const float*)d_in[4];
    const float* fB2 = (const float*)d_in[5];
    const float* dW1 = (const float*)d_in[14];
    const float* dB1 = (const float*)d_in[15];
    const float* dW2 = (const float*)d_in[16];
    const float* dB2 = (const float*)d_in[17];
    const int*   adj_rows = (const int*)d_in[26];
    const int*   adj_cols = (const int*)d_in[27];
    const float* adj_vals = (const float*)d_in[28];
    const int*   ufeat    = (const int*)d_in[29];
    const int*   ubatch   = (const int*)d_in[30];
    const float* rbatch   = (const float*)d_in[31];
    const int*   ibatch   = (const int*)d_in[32];
    float* out = (float*)d_out;

    float* user_f0     = (float*)d_ws;
    float* item_f0     = user_f0 + (long long)NUM_U * 64;
    float* users_local = item_f0 + (long long)NUM_I * 64;
    float* acc         = users_local + (long long)NUM_U * 64;   // 16 floats

    // 1. zero SpMM accumulator + scalar accumulators (contiguous)
    long long zn = (long long)NUM_U * 64 + 16;
    zero_kernel<<<(int)((zn + 255) / 256), 256, 0, stream>>>(users_local, zn);

    // 2-3. fused filter0 MLP forward (WMMA f16)
    filter_fwd_kernel<<<(NUM_U + 63) / 64, 128, 0, stream>>>(
        user_emb, NUM_U, fW1, fB1, fW2, fB2, user_f0);
    filter_fwd_kernel<<<(NUM_I + 63) / 64, 128, 0, stream>>>(
        item_emb, NUM_I, fW1, fB1, fW2, fB2, item_f0);

    // 4. disc0 CE loss on filtered users -> acc[0]
    disc_loss_kernel<<<(NUM_U + 63) / 64, 128, 0, stream>>>(
        user_f0, NUM_U, dW1, dB1, dW2, dB2, ufeat, acc + 0);

    // 5. SpMM: users_local = A @ item_f0
    long long spthreads = (long long)NNZE * 32;
    spmm_kernel<<<(int)((spthreads + 255) / 256), 256, 0, stream>>>(
        adj_rows, adj_cols, adj_vals, item_f0, users_local);

    // 6. disc0 CE loss on aggregated users -> acc[1]
    disc_loss_kernel<<<(NUM_U + 63) / 64, 128, 0, stream>>>(
        users_local, NUM_U, dW1, dB1, dW2, dB2, ufeat, acc + 1);

    // 7. rating prediction branch -> acc[2], acc[3]
    pred_kernel<<<(BATCH * 32 + 255) / 256, 256, 0, stream>>>(
        user_f0, item_f0, ubatch, ibatch, rbatch, acc);

    // 8. combine into the 3 output scalars
    finalize_kernel<<<1, 64, 0, stream>>>(acc, out);
}
